// my_CRF_30674656428164
// MI455X (gfx1250) — matrix operand order repaired and checked by reading, not compiled, not run on hardware
//
#include <hip/hip_runtime.h>

#define Bn 256
#define Sn 1024
#define Tn 128

typedef __attribute__((ext_vector_type(16))) _Float16 v16h;
typedef __attribute__((ext_vector_type(8)))  float    v8f;
typedef __attribute__((ext_vector_type(2)))  _Float16 h2t;

#if defined(__has_builtin)
# if __has_builtin(__builtin_amdgcn_global_load_async_to_lds_b128) && \
     __has_builtin(__builtin_amdgcn_s_wait_asynccnt)
#  define USE_ASYNC_LDS 1
# endif
#endif

#if defined(USE_ASYNC_LDS)
// Parameter types per hipcc diagnostic: pointer to 16-byte int vector,
// global (AS1) source and LDS (AS3) destination.
typedef int vi4 __attribute__((vector_size(4 * sizeof(int))));
typedef __attribute__((address_space(1))) vi4 gvi4;
typedef __attribute__((address_space(3))) vi4 lvi4;

// global -> LDS async DMA of 16 bytes per lane (ASYNCcnt-tracked).
__device__ __forceinline__ void async_copy16(const float* g, float* l) {
  __builtin_amdgcn_global_load_async_to_lds_b128(
      (gvi4*)(uintptr_t)g, (lvi4*)(uintptr_t)l, 0, 0);
}
#endif

// ---------------------------------------------------------------- zero ws
__global__ void crf_zero(float* gold) { gold[threadIdx.x] = 0.0f; }

// ---------------------------------------------------------------- gold score
// grid (B, S/8), 256 threads = 8 waves, one wave per sequence position.
__global__ void __launch_bounds__(256) crf_gold(
    const float* __restrict__ em, const int* __restrict__ tags,
    const float* __restrict__ mask, const float* __restrict__ startT,
    const float* __restrict__ endT, const float* __restrict__ trans,
    float* __restrict__ gold)
{
  __shared__ float acc;
  const int b = blockIdx.x;
  const int wave = threadIdx.x >> 5, lane = threadIdx.x & 31;
  const int s = blockIdx.y * 8 + wave;
  if (threadIdx.x == 0) acc = 0.0f;
  __syncthreads();

  const float4 v = ((const float4*)(em + ((size_t)b * Sn + s) * Tn))[lane];
  float mx = fmaxf(fmaxf(v.x, v.y), fmaxf(v.z, v.w));
  #pragma unroll
  for (int off = 16; off > 0; off >>= 1) mx = fmaxf(mx, __shfl_xor(mx, off, 32));
  float sum = __expf(v.x - mx) + __expf(v.y - mx) + __expf(v.z - mx) + __expf(v.w - mx);
  #pragma unroll
  for (int off = 16; off > 0; off >>= 1) sum += __shfl_xor(sum, off, 32);
  const float lse = mx + __logf(sum);

  const int tag = tags[b * Sn + s];
  const float cand = (tag & 2) ? ((tag & 1) ? v.w : v.z) : ((tag & 1) ? v.y : v.x);
  const float et = __shfl(cand, tag >> 2, 32);

  if (lane == 0) {
    const float ms = mask[b * Sn + s];
    float contrib = (et - lse) * ms;                      // emission log-prob
    if (s > 0) contrib += trans[tags[b * Sn + s - 1] * Tn + tag] * ms;
    else       contrib += startT[tag];
    if (s == Sn - 1) contrib += endT[tag] * ms;
    atomicAdd(&acc, contrib);
  }
  __syncthreads();
  if (threadIdx.x == 0) atomicAdd(&gold[b], acc);
}

// ---------------------------------------------------------------- forward recursion
// One workgroup per 16 batches; 8 waves, wave w owns output columns [16w,16w+16).
// Per step:  S = P(alpha) @ Q(trans)  via v_wmma_f32_16x16x32_f16, K=128 (4 WMMAs/wave).
// Q fragments (step-invariant) live permanently in VGPRs; emissions double-buffered
// in LDS (async-to-LDS DMA when the toolchain exposes it).
__global__ void __launch_bounds__(256, 1) crf_forward(
    const float* __restrict__ em, const float* __restrict__ mask,
    const float* __restrict__ startT, const float* __restrict__ endT,
    const float* __restrict__ trans, float* __restrict__ fwd)
{
  __shared__ _Float16 Qs[Tn][Tn];        // [n][k] = exp(trans[k][n]-cmax[n])  (32 KB)
  __shared__ float    tcolmax[Tn];       // cmax[n]
  __shared__ _Float16 Pa[16][Tn];        // P[m][k] = exp(alpha-amax)          (4 KB)
  __shared__ float    alphaF[16][Tn];    // f32 alpha                          (8 KB)
  __shared__ float    emb[2][16][Tn];    // double-buffered emissions          (16 KB)
  __shared__ float    amax[16];
  __shared__ float    rowmax[16][8];

  const int t = threadIdx.x, wave = t >> 5, lane = t & 31;
  const int half_ = lane >> 4, l15 = lane & 15;
  const int bb = blockIdx.x * 16;

  // ---- column maxes of transitions
  if (t < Tn) {
    float mxv = -3.0e38f;
    for (int i = 0; i < Tn; ++i) mxv = fmaxf(mxv, trans[i * Tn + t]);
    tcolmax[t] = mxv;
  }
  __syncthreads();
  // ---- Qs[n][k] (coalesced global read, transposed LDS write; done once)
  for (int idx = t; idx < Tn * Tn; idx += 256) {
    const int k = idx >> 7, n = idx & 127;
    Qs[n][k] = (_Float16)__expf(trans[k * Tn + n] - tcolmax[n]);
  }
  // ---- alpha0 = start + emissions[:,0]
  for (int idx = t; idx < 16 * Tn; idx += 256) {
    const int m = idx >> 7, j = idx & 127;
    alphaF[m][j] = startT[j] + em[(((size_t)(bb + m)) * Sn) * Tn + j];
  }
  __syncthreads();
  if (t < 16) {
    float mxv = -3.0e38f;
    for (int j = 0; j < Tn; ++j) mxv = fmaxf(mxv, alphaF[t][j]);
    amax[t] = mxv;
  }
  __syncthreads();
  for (int idx = t; idx < 16 * Tn; idx += 256) {
    const int m = idx >> 7, j = idx & 127;
    Pa[m][j] = (_Float16)__expf(alphaF[m][j] - amax[m]);
  }
  // ---- stage emissions for s=1 into emb[1]
  {
    const int m = t >> 4, j0 = (t & 15) * 8;
    const float* src = em + (((size_t)(bb + m)) * Sn + 1) * Tn + j0;
#if defined(USE_ASYNC_LDS)
    async_copy16(src,     &emb[1][m][j0]);
    async_copy16(src + 4, &emb[1][m][j0 + 4]);
    __builtin_amdgcn_s_wait_asynccnt(0);
#else
    ((float4*)&emb[1][m][j0])[0] = ((const float4*)src)[0];
    ((float4*)&emb[1][m][j0])[1] = ((const float4*)src)[1];
#endif
  }
  __syncthreads();

  // ---- persistent B fragments (lane half h holds k = 16h + s, contiguous; same order as A)
  v16h bfrag[4];
  {
    const int ncol = wave * 16 + l15;
    #pragma unroll
    for (int kb = 0; kb < 4; ++kb) {
      v16h bf;
      #pragma unroll
      for (int r = 0; r < 8; ++r) {
        const int k = kb * 32 + 2 * r + 16 * half_;
        const h2t q = *(const h2t*)&Qs[ncol][k];
        bf[2 * r] = q.x; bf[2 * r + 1] = q.y;
      }
      bfrag[kb] = bf;
    }
  }

  const int ncol0 = wave * 16;
  for (int s = 1; s < Sn; ++s) {
    const int cur = s & 1, nxt = cur ^ 1;
    const bool have = (s + 1 < Sn);
    const int mld = t >> 4, j0 = (t & 15) * 8;
    const float* src = em + (((size_t)(bb + mld)) * Sn + s + 1) * Tn + j0;

#if defined(USE_ASYNC_LDS)
    if (have) {                           // async DMA next step's tile into emb[nxt]
      async_copy16(src,     &emb[nxt][mld][j0]);
      async_copy16(src + 4, &emb[nxt][mld][j0 + 4]);
      if (s + 2 < Sn) __builtin_prefetch(src + Tn, 0, 1);
    }
#else
    float4 e0, e1;
    if (have) {
      e0 = ((const float4*)src)[0];
      e1 = ((const float4*)src)[1];
      if (s + 2 < Sn) __builtin_prefetch(src + Tn, 0, 1);
    }
#endif

    // ---- P @ Q  (A fragments: lane half h holds k = 16h + s, same order as B)
    v8f c = {};
    #pragma unroll
    for (int kb = 0; kb < 4; ++kb) {
      v16h a;
      #pragma unroll
      for (int r = 0; r < 8; ++r) {
        const int koff = 2 * r + 16 * half_;
        const h2t p = *(const h2t*)&Pa[l15][kb * 32 + koff];
        a[2 * r] = p.x; a[2 * r + 1] = p.y;
      }
      c = __builtin_amdgcn_wmma_f32_16x16x32_f16(false, a, false, bfrag[kb],
                                                 (short)0, c, false, false);
    }

    // ---- new alpha + per-row tile max (C/D layout: n=lane&15, m=r+8*(lane>>4))
    float vals[8];
    #pragma unroll
    for (int r = 0; r < 8; ++r) {
      const int m = r + 8 * half_;
      const int n = ncol0 + l15;
      const float val = emb[cur][m][n] + amax[m] + tcolmax[n] + __logf(c[r]);
      vals[r] = val;
      alphaF[m][n] = val;
      float rmx = val;
      #pragma unroll
      for (int off = 1; off < 16; off <<= 1) rmx = fmaxf(rmx, __shfl_xor(rmx, off, 16));
      if (l15 == 0) rowmax[m][wave] = rmx;
    }
    __syncthreads();                                   // rowmax/alphaF visible
    if (t < 16) {
      float mxv = rowmax[t][0];
      #pragma unroll
      for (int w = 1; w < 8; ++w) mxv = fmaxf(mxv, rowmax[t][w]);
      amax[t] = mxv;
    }
#if !defined(USE_ASYNC_LDS)
    if (have) {   // land prefetched emissions into the other buffer
      ((float4*)&emb[nxt][mld][j0])[0] = e0;
      ((float4*)&emb[nxt][mld][j0])[1] = e1;
    }
#endif
    __syncthreads();                                   // amax ready
    // ---- next-step P directly from registers (no extra LDS pass)
    #pragma unroll
    for (int r = 0; r < 8; ++r) {
      const int m = r + 8 * half_;
      Pa[m][ncol0 + l15] = (_Float16)__expf(vals[r] - amax[m]);
    }
#if defined(USE_ASYNC_LDS)
    if (have) __builtin_amdgcn_s_wait_asynccnt(0);     // emb[nxt] landed
#endif
    __syncthreads();                                   // Pa (and emb) ready
  }

  // ---- fwd[b] = logsumexp(alpha*mask[:,S-1] + end)
  if (t < 16) {
    const int m = t;
    const float msk = mask[((size_t)(bb + m)) * Sn + (Sn - 1)];
    float mxv = -3.0e38f;
    for (int j = 0; j < Tn; ++j) mxv = fmaxf(mxv, alphaF[m][j] * msk + endT[j]);
    float sum = 0.0f;
    for (int j = 0; j < Tn; ++j) sum += __expf(alphaF[m][j] * msk + endT[j] - mxv);
    fwd[bb + m] = mxv + __logf(sum);
  }
}

// ---------------------------------------------------------------- mean(fwd - gold)
__global__ void crf_final(const float* __restrict__ fwd, const float* __restrict__ gold,
                          float* __restrict__ out)
{
  __shared__ float red[Bn];
  const int t = threadIdx.x;
  red[t] = fwd[t] - gold[t];
  __syncthreads();
  for (int off = Bn / 2; off > 0; off >>= 1) {
    if (t < off) red[t] += red[t + off];
    __syncthreads();
  }
  if (t == 0) out[0] = red[0] * (1.0f / Bn);
}

extern "C" void kernel_launch(void* const* d_in, const int* in_sizes, int n_in,
                              void* d_out, int out_size, void* d_ws, size_t ws_size,
                              hipStream_t stream)
{
  (void)in_sizes; (void)n_in; (void)out_size; (void)ws_size;
  const float* em     = (const float*)d_in[0];
  const int*   tags   = (const int*)d_in[1];
  const float* mask   = (const float*)d_in[2];
  const float* startT = (const float*)d_in[3];
  const float* endT   = (const float*)d_in[4];
  const float* trans  = (const float*)d_in[5];

  float* gold = (float*)d_ws;
  float* fwd  = gold + Bn;
  float* out  = (float*)d_out;

  crf_zero<<<1, Bn, 0, stream>>>(gold);
  crf_gold<<<dim3(Bn, Sn / 8), 256, 0, stream>>>(em, tags, mask, startT, endT, trans, gold);
  crf_forward<<<Bn / 16, 256, 0, stream>>>(em, mask, startT, endT, trans, fwd);
  crf_final<<<1, Bn, 0, stream>>>(fwd, gold, out);
}
